// Transformer_64252710748526
// MI455X (gfx1250) — compile-verified, matrix operand order checked by reading, not run on hardware
//
#include <hip/hip_runtime.h>
#include <hip/hip_bf16.h>
#include <math.h>

typedef __bf16 bf16;
typedef __bf16 v16bf __attribute__((ext_vector_type(16)));
typedef float  v8f   __attribute__((ext_vector_type(8)));
typedef int    v4i   __attribute__((ext_vector_type(4)));

union AF { v16bf v; uint4 u[2]; };

#define BN   4096      // B*N rows
#define NSEQ 2048
#define NB_B 2
#define DIMC 1024
#define HDS  8
#define DH   64
#define INNERC 512
#define KVLD 128
#define FFI  2730
#define FF2K 2752      // FFI padded to multiple of 32 (K pad for FF2 GEMM)
#define FFN  5460      // 2*FFI
#define FFNP 5504      // FFN padded to multiple of 64 (N pad for FF1 GEMM)

// gfx1250 async Global->LDS staging (ASYNCcnt path), with safe fallback.
#if defined(__AMDGCN__) && __has_builtin(__builtin_amdgcn_global_load_async_to_lds_b128)
#define USE_ASYNC 1
#define AS1 __attribute__((address_space(1)))
#define AS3 __attribute__((address_space(3)))
#if __has_builtin(__builtin_amdgcn_s_wait_asynccnt)
#define WAIT_ASYNC() __builtin_amdgcn_s_wait_asynccnt(0)
#else
#define WAIT_ASYNC() asm volatile("s_wait_asynccnt 0" ::: "memory")
#endif
#else
#define USE_ASYNC 0
#endif

// ---------------------------------------------------------------------------
// f32 [K][Nout] row-major -> bf16 [Noutp][Kpad] (K-contiguous per output col,
// zero-padded in both K (to Kpad) and N (to Noutp) so GEMM needs no guards).
// ---------------------------------------------------------------------------
__global__ void convert_w_kernel(const float* __restrict__ w, bf16* __restrict__ wt,
                                 int K, int Nout, int Kpad, int Noutp) {
    size_t idx = (size_t)blockIdx.x * 256 + threadIdx.x;
    size_t total = (size_t)Noutp * Kpad;
    if (idx >= total) return;
    int k = (int)(idx % Kpad);
    int n = (int)(idx / Kpad);
    float v = (k < K && n < Nout) ? w[(size_t)k * Nout + n] : 0.0f;
    wt[idx] = (bf16)v;
}

// ---------------------------------------------------------------------------
// LayerNorm over DIM=1024; one row per 256-thread block (8 waves).
// outB != nullptr -> bf16 output; else f32 output (final LN).
// ---------------------------------------------------------------------------
__global__ void ln_kernel(const float* __restrict__ x, const float* __restrict__ g,
                          const float* __restrict__ bta, bf16* __restrict__ outB,
                          float* __restrict__ outF) {
    int row = blockIdx.x, t = threadIdx.x;
    const float* xr = x + (size_t)row * DIMC;
    float v[4]; float s = 0.f, s2 = 0.f;
    #pragma unroll
    for (int i = 0; i < 4; i++) { float xv = xr[t + i * 256]; v[i] = xv; s += xv; s2 += xv * xv; }
    __shared__ float red1[8], red2[8];
    int lane = t & 31, w = t >> 5;
    #pragma unroll
    for (int off = 16; off; off >>= 1) { s += __shfl_down(s, off, 32); s2 += __shfl_down(s2, off, 32); }
    if (lane == 0) { red1[w] = s; red2[w] = s2; }
    __syncthreads();
    float tot = 0.f, tot2 = 0.f;
    #pragma unroll
    for (int i = 0; i < 8; i++) { tot += red1[i]; tot2 += red2[i]; }
    float mu = tot * (1.0f / DIMC);
    float var = tot2 * (1.0f / DIMC) - mu * mu;
    float rstd = rsqrtf(var + 1e-5f);
    #pragma unroll
    for (int i = 0; i < 4; i++) {
        int col = t + i * 256;
        float y = (v[i] - mu) * rstd * g[col] + bta[col];
        if (outB) outB[(size_t)row * DIMC + col] = (bf16)y;
        else      outF[(size_t)row * DIMC + col] = y;
    }
}

// ---------------------------------------------------------------------------
// bf16 GEMM:  C[M x Nout] = A[M x K] (row-major, lda) * Wt[Noutp x Kpad]^T
// 256 threads = 8 waves (4 M-slabs x 2 N-slabs); block tile 128x64,
// wave tile 32x32 = 2x2 v_wmma_f32_16x16x32_bf16 accumulators, K step 32.
// Staging: async Global->LDS b128 (ASYNCcnt) when available.
// mode 0: store bf16 (ldc, guarded by Nout). mode 1: outF = resid + C (f32).
// ---------------------------------------------------------------------------
__global__ __launch_bounds__(256)
void gemm_kernel(const bf16* __restrict__ A, int lda,
                 const bf16* __restrict__ Wt, int ldb,
                 int Nout, int K,
                 bf16* __restrict__ outB, float* __restrict__ outF,
                 const float* __restrict__ resid, int ldc, int mode) {
    __shared__ alignas(16) bf16 As[128][32];
    __shared__ alignas(16) bf16 Bs[64][32];
    int t = threadIdx.x;
    int lane = t & 31, wv = t >> 5;
    int half = lane >> 4, l16 = lane & 15;
    int waveM = wv >> 1, waveN = wv & 1;
    int m0 = blockIdx.y * 128;
    int n0 = blockIdx.x * 64;
    int r0 = t >> 2, q0 = t & 3;   // staging coords: row, 8-elem segment

    v8f acc[2][2] = {};
    for (int k0 = 0; k0 < K; k0 += 32) {
#if USE_ASYNC
        // async DMA: global -> LDS, no VGPR data movement (ASYNCcnt-tracked)
        __builtin_amdgcn_global_load_async_to_lds_b128(
            (AS1 v4i*)(void*)&A[(size_t)(m0 + r0) * lda + k0 + q0 * 8],
            (AS3 v4i*)(void*)&As[r0][q0 * 8], 0, 0);
        __builtin_amdgcn_global_load_async_to_lds_b128(
            (AS1 v4i*)(void*)&A[(size_t)(m0 + r0 + 64) * lda + k0 + q0 * 8],
            (AS3 v4i*)(void*)&As[r0 + 64][q0 * 8], 0, 0);
        __builtin_amdgcn_global_load_async_to_lds_b128(
            (AS1 v4i*)(void*)&Wt[(size_t)(n0 + r0) * ldb + k0 + q0 * 8],
            (AS3 v4i*)(void*)&Bs[r0][q0 * 8], 0, 0);
        WAIT_ASYNC();
#else
        *(uint4*)&As[r0][q0 * 8] =
            *(const uint4*)&A[(size_t)(m0 + r0) * lda + k0 + q0 * 8];
        *(uint4*)&As[r0 + 64][q0 * 8] =
            *(const uint4*)&A[(size_t)(m0 + r0 + 64) * lda + k0 + q0 * 8];
        *(uint4*)&Bs[r0][q0 * 8] =
            *(const uint4*)&Wt[(size_t)(n0 + r0) * ldb + k0 + q0 * 8];
#endif
        __syncthreads();
        AF af[2], bfm[2];
        #pragma unroll
        for (int i = 0; i < 2; i++) {
            int r = waveM * 32 + i * 16 + l16;
            af[i].u[0] = *(const uint4*)&As[r][half * 8];
            af[i].u[1] = *(const uint4*)&As[r][half * 8 + 16];
        }
        #pragma unroll
        for (int j = 0; j < 2; j++) {
            int r = waveN * 32 + j * 16 + l16;
            bfm[j].u[0] = *(const uint4*)&Bs[r][half * 8];
            bfm[j].u[1] = *(const uint4*)&Bs[r][half * 8 + 16];
        }
        #pragma unroll
        for (int i = 0; i < 2; i++)
            #pragma unroll
            for (int j = 0; j < 2; j++)
                acc[i][j] = __builtin_amdgcn_wmma_f32_16x16x32_bf16(
                    false, af[i].v, false, bfm[j].v, (short)0, acc[i][j], false, false);
        __syncthreads();
    }
    // epilogue: C layout row = r + 8*half (per VGPR), col = lane&15
    #pragma unroll
    for (int i = 0; i < 2; i++)
        #pragma unroll
        for (int j = 0; j < 2; j++) {
            int n = n0 + waveN * 32 + j * 16 + l16;
            if (n >= Nout) continue;   // only FF1 GEMM has ragged Nout
            #pragma unroll
            for (int r = 0; r < 8; r++) {
                int m = m0 + waveM * 32 + i * 16 + r + 8 * half;
                float val = acc[i][j][r];
                if (mode == 0) outB[(size_t)m * ldc + n] = (bf16)val;
                else outF[(size_t)m * ldc + n] = resid[(size_t)m * ldc + n] + val;
            }
        }
}

// ---------------------------------------------------------------------------
// Flash causal multi-query attention with T5 rel-pos bias.
// grid = (N/16, B); 256 threads = 8 waves, wave w = head w.
// Each wave: 16 query rows x DH=64, streams keys in blocks of 32.
// Q@K^T and P@V via v_wmma_f32_16x16x32_bf16.
// ---------------------------------------------------------------------------
__global__ __launch_bounds__(256)
void attn_kernel(const bf16* __restrict__ q, const bf16* __restrict__ kv,
                 const float* __restrict__ rel_emb, bf16* __restrict__ o) {
    int t = threadIdx.x, lane = t & 31, h = t >> 5;
    int half = lane >> 4, l16 = lane & 15;
    int qstart = blockIdx.x * 16;
    int b = blockIdx.y;

    __shared__ alignas(16) bf16 vt[64][32];      // V tile transposed [dh][key]
    __shared__ alignas(16) bf16 pbuf[8][16][32]; // per-wave P bounce buffer
    __shared__ float biasTab[8][32];             // per-head rel-pos table

    biasTab[h][lane] = rel_emb[(size_t)lane * HDS + h];  // NB=32 == wave32

    // Q fragments (A layout): q row-major, dh contiguous
    AF qa[2];
    {
        int row = qstart + l16;
        const bf16* qp = q + (size_t)(b * NSEQ + row) * INNERC + h * DH;
        #pragma unroll
        for (int c = 0; c < 2; c++) {
            qa[c].u[0] = *(const uint4*)(qp + c * 32 + half * 8);
            qa[c].u[1] = *(const uint4*)(qp + c * 32 + half * 8 + 16);
        }
    }

    v8f oc[4] = {};
    float mrow[8], lrow[8];
    #pragma unroll
    for (int r = 0; r < 8; r++) { mrow[r] = -3.0e38f; lrow[r] = 0.f; }
    const float scale = 0.125f;             // DH^-0.5
    const float blog = 0.480898346f * 16.f; // 16/log(128/16)

    int kend = qstart + 16;
    for (int ks = 0; ks < kend; ks += 32) {
        __syncthreads();
        // cooperative V staging (shared across all 8 heads: multi-query)
        for (int idx = t; idx < 64 * 32; idx += 256) {
            int dh = idx & 63, key = idx >> 6;
            vt[dh][key] = kv[(size_t)(b * NSEQ + ks + key) * KVLD + DH + dh];
        }
        __syncthreads();

        float p[2][8];
        #pragma unroll
        for (int tt = 0; tt < 2; tt++) {
            int j0 = ks + tt * 16;
            // K^T fragments (B layout): lane col = key, dh contiguous
            AF kb[2];
            const bf16* kp = kv + (size_t)(b * NSEQ + j0 + l16) * KVLD;
            #pragma unroll
            for (int c = 0; c < 2; c++) {
                kb[c].u[0] = *(const uint4*)(kp + c * 32 + half * 8);
                kb[c].u[1] = *(const uint4*)(kp + c * 32 + half * 8 + 16);
            }
            v8f s = {};
            s = __builtin_amdgcn_wmma_f32_16x16x32_bf16(false, qa[0].v, false, kb[0].v, (short)0, s, false, false);
            s = __builtin_amdgcn_wmma_f32_16x16x32_bf16(false, qa[1].v, false, kb[1].v, (short)0, s, false, false);
            int j = j0 + l16;
            #pragma unroll
            for (int r = 0; r < 8; r++) {
                int i = qstart + r + 8 * half;
                float lg;
                if (j > i) {
                    lg = -3.0e38f;
                } else {
                    int dist = i - j;
                    int bucket = dist;
                    if (dist >= 16) {
                        bucket = 16 + (int)(logf((float)dist * (1.0f / 16.0f)) * blog);
                        if (bucket > 31) bucket = 31;
                    }
                    lg = s[r] * scale + biasTab[h][bucket];
                }
                p[tt][r] = lg;
            }
        }
        // online softmax: row data lives across a 16-lane group (same half)
        #pragma unroll
        for (int r = 0; r < 8; r++) {
            float rm = fmaxf(p[0][r], p[1][r]);
            #pragma unroll
            for (int off = 8; off; off >>= 1) rm = fmaxf(rm, __shfl_xor(rm, off, 32));
            float mn = fmaxf(mrow[r], rm);
            float corr = __expf(mrow[r] - mn);
            float e0 = __expf(p[0][r] - mn);
            float e1 = __expf(p[1][r] - mn);
            p[0][r] = e0; p[1][r] = e1;
            float rs = e0 + e1;
            #pragma unroll
            for (int off = 8; off; off >>= 1) rs += __shfl_xor(rs, off, 32);
            lrow[r] = lrow[r] * corr + rs;
            mrow[r] = mn;
            #pragma unroll
            for (int c = 0; c < 4; c++) oc[c][r] *= corr;
        }
        // C-layout -> A-layout via per-wave LDS bounce (in-order DS, no barrier)
        #pragma unroll
        for (int r = 0; r < 8; r++) {
            int row = r + 8 * half;
            pbuf[h][row][l16]      = (bf16)p[0][r];
            pbuf[h][row][16 + l16] = (bf16)p[1][r];
        }
        AF pa;
        pa.u[0] = *(const uint4*)&pbuf[h][l16][half * 8];
        pa.u[1] = *(const uint4*)&pbuf[h][l16][half * 8 + 16];
        #pragma unroll
        for (int c = 0; c < 4; c++) {
            AF vb;
            int dh = c * 16 + l16;
            vb.u[0] = *(const uint4*)&vt[dh][half * 8];
            vb.u[1] = *(const uint4*)&vt[dh][half * 8 + 16];
            oc[c] = __builtin_amdgcn_wmma_f32_16x16x32_bf16(
                false, pa.v, false, vb.v, (short)0, oc[c], false, false);
        }
    }
    // normalize + store
    #pragma unroll
    for (int c = 0; c < 4; c++) {
        int dh = c * 16 + l16;
        #pragma unroll
        for (int r = 0; r < 8; r++) {
            int row = qstart + r + 8 * half;
            float val = oc[c][r] / lrow[r];
            o[(size_t)(b * NSEQ + row) * INNERC + h * DH + dh] = (bf16)val;
        }
    }
}

// ---------------------------------------------------------------------------
// GEGLU: act[m][j] = gelu_exact(h[m][FFI+j]) * h[m][j], zero-padded to FF2K.
// ---------------------------------------------------------------------------
__global__ void geglu_kernel(const bf16* __restrict__ hbuf, bf16* __restrict__ act) {
    size_t idx = (size_t)blockIdx.x * 256 + threadIdx.x;
    const size_t total = (size_t)BN * FF2K;
    if (idx >= total) return;
    int j = (int)(idx % FF2K);
    int m = (int)(idx / FF2K);
    float outv = 0.f;
    if (j < FFI) {
        float a = (float)hbuf[(size_t)m * FFN + j];
        float g = (float)hbuf[(size_t)m * FFN + FFI + j];
        float gelu = 0.5f * g * (1.0f + erff(g * 0.70710678f));
        outv = gelu * a;
    }
    act[idx] = (bf16)outv;
}

// ---------------------------------------------------------------------------
extern "C" void kernel_launch(void* const* d_in, const int* in_sizes, int n_in,
                              void* d_out, int out_size, void* d_ws, size_t ws_size,
                              hipStream_t stream) {
    (void)in_sizes; (void)n_in; (void)out_size; (void)ws_size;
    const float* in_x  = (const float*)d_in[0];
    const float* rel_e = (const float*)d_in[1];
    const float* ln1_g = (const float*)d_in[2];
    const float* ln1_b = (const float*)d_in[3];
    const float* wq    = (const float*)d_in[4];
    const float* wkv   = (const float*)d_in[5];
    const float* wo    = (const float*)d_in[6];
    const float* ln2_g = (const float*)d_in[7];
    const float* ln2_b = (const float*)d_in[8];
    const float* w1    = (const float*)d_in[9];
    const float* w2    = (const float*)d_in[10];
    const float* lnf_g = (const float*)d_in[11];
    const float* lnf_b = (const float*)d_in[12];
    float* out = (float*)d_out;

    char* ws = (char*)d_ws;
    size_t off = 0;
    auto alloc = [&](size_t bytes) -> void* {
        off = (off + 255) & ~(size_t)255;
        void* p = ws + off;
        off += bytes;
        return p;
    };
    float* xbuf  = (float*)alloc((size_t)BN * DIMC * 4);
    bf16*  xn    = (bf16*) alloc((size_t)BN * DIMC * 2);
    bf16*  qb    = (bf16*) alloc((size_t)BN * INNERC * 2);
    bf16*  kvb   = (bf16*) alloc((size_t)BN * KVLD * 2);
    bf16*  ob    = (bf16*) alloc((size_t)BN * INNERC * 2);
    bf16*  hb    = (bf16*) alloc((size_t)BN * FFN * 2);
    bf16*  actb  = (bf16*) alloc((size_t)BN * FF2K * 2);
    bf16*  wq_t  = (bf16*) alloc((size_t)INNERC * DIMC * 2);
    bf16*  wkv_t = (bf16*) alloc((size_t)KVLD * DIMC * 2);
    bf16*  wo_t  = (bf16*) alloc((size_t)DIMC * INNERC * 2);
    bf16*  w1_t  = (bf16*) alloc((size_t)FFNP * DIMC * 2);
    bf16*  w2_t  = (bf16*) alloc((size_t)DIMC * FF2K * 2);

    // residual stream (grad_shrink forward == identity)
    (void)hipMemcpyAsync(xbuf, in_x, (size_t)BN * DIMC * 4, hipMemcpyDeviceToDevice, stream);

    dim3 blk(256);
    for (int l = 0; l < 6; l++) {
        const float* wq_l  = wq  + (size_t)l * DIMC * INNERC;
        const float* wkv_l = wkv + (size_t)l * DIMC * (2 * DH);
        const float* wo_l  = wo  + (size_t)l * INNERC * DIMC;
        const float* w1_l  = w1  + (size_t)l * DIMC * FFN;
        const float* w2_l  = w2  + (size_t)l * FFI * DIMC;

        // convert this layer's weights to bf16 [Noutp][Kpad]
        convert_w_kernel<<<((size_t)INNERC * DIMC + 255) / 256, blk, 0, stream>>>(wq_l,  wq_t,  DIMC, INNERC, DIMC, INNERC);
        convert_w_kernel<<<((size_t)KVLD   * DIMC + 255) / 256, blk, 0, stream>>>(wkv_l, wkv_t, DIMC, KVLD,   DIMC, KVLD);
        convert_w_kernel<<<((size_t)DIMC * INNERC + 255) / 256, blk, 0, stream>>>(wo_l,  wo_t,  INNERC, DIMC, INNERC, DIMC);
        convert_w_kernel<<<((size_t)FFNP * DIMC   + 255) / 256, blk, 0, stream>>>(w1_l,  w1_t,  DIMC, FFN,    DIMC, FFNP);
        convert_w_kernel<<<((size_t)DIMC * FF2K   + 255) / 256, blk, 0, stream>>>(w2_l,  w2_t,  FFI,  DIMC,   FF2K, DIMC);

        // --- attention block ---
        ln_kernel<<<BN, blk, 0, stream>>>(xbuf, ln1_g + l * DIMC, ln1_b + l * DIMC, xn, nullptr);
        gemm_kernel<<<dim3(INNERC / 64, BN / 128), blk, 0, stream>>>(
            xn, DIMC, wq_t, DIMC, INNERC, DIMC, qb, nullptr, nullptr, INNERC, 0);
        gemm_kernel<<<dim3(KVLD / 64, BN / 128), blk, 0, stream>>>(
            xn, DIMC, wkv_t, DIMC, KVLD, DIMC, kvb, nullptr, nullptr, KVLD, 0);
        attn_kernel<<<dim3(NSEQ / 16, NB_B), blk, 0, stream>>>(qb, kvb, rel_e, ob);
        gemm_kernel<<<dim3(DIMC / 64, BN / 128), blk, 0, stream>>>(
            ob, INNERC, wo_t, INNERC, DIMC, INNERC, nullptr, xbuf, xbuf, DIMC, 1);

        // --- GEGLU feedforward ---
        ln_kernel<<<BN, blk, 0, stream>>>(xbuf, ln2_g + l * DIMC, ln2_b + l * DIMC, xn, nullptr);
        gemm_kernel<<<dim3(FFNP / 64, BN / 128), blk, 0, stream>>>(
            xn, DIMC, w1_t, DIMC, FFN, DIMC, hb, nullptr, nullptr, FFN, 0);
        geglu_kernel<<<((size_t)BN * FF2K + 255) / 256, blk, 0, stream>>>(hb, actb);
        gemm_kernel<<<dim3(DIMC / 64, BN / 128), blk, 0, stream>>>(
            actb, FF2K, w2_t, FF2K, DIMC, FF2K, nullptr, xbuf, xbuf, DIMC, 1);
    }
    // final LN -> f32 output
    ln_kernel<<<BN, blk, 0, stream>>>(xbuf, lnf_g, lnf_b, nullptr, out);
}
